// MultiHeadSelfAttention_76390288327390
// MI455X (gfx1250) — matrix operand order repaired and checked by reading, not compiled, not run on hardware
//
#include <hip/hip_runtime.h>

// MHA: B=2, T=4096, D=512, H=8, DK=64 — bf16 WMMA (f32 accum) flash attention.

typedef __bf16 bf16;
typedef __attribute__((ext_vector_type(8)))  bf16  v8bf;
typedef __attribute__((ext_vector_type(16))) bf16  v16bf;
typedef __attribute__((ext_vector_type(8)))  float v8f;

#define TB 2
#define TT 4096
#define TD 512
#define TH 8
#define TDK 64

__device__ __forceinline__ v16bf cat16(v8bf lo, v8bf hi) {
  return __builtin_shufflevector(lo, hi, 0,1,2,3,4,5,6,7,8,9,10,11,12,13,14,15);
}

// A-fragment (16x32, MxK) from row-major [*, ld]; p points at (row 0, kbase).
// Lane l: m = l&15, h = l>>4. elems 0..7 = K[8h..8h+8), elems 8..15 = K[16+8h..).
__device__ __forceinline__ v16bf load_a(const bf16* p, int ld, int m, int h) {
  const bf16* q = p + (size_t)m * ld + 8 * h;
  v8bf lo = *(const v8bf*)q;
  v8bf hi = *(const v8bf*)(q + 16);
  return cat16(lo, hi);
}

// B-fragment (32x16, KxN) from row-major [K, ld]; p points at (kbase, nbase).
// Lane l holds N-col (l&15); elems 0..7 = K[8h..8h+8), elems 8..15 = K[16+8h..).
__device__ __forceinline__ v16bf load_b(const bf16* p, int ld, int m, int h) {
  v16bf r;
#pragma unroll
  for (int s = 0; s < 8; ++s) {
    r[s]     = p[(size_t)(8 * h + s) * ld + m];
    r[8 + s] = p[(size_t)(16 + 8 * h + s) * ld + m];
  }
  return r;
}

__device__ __forceinline__ v8f wmma_bf16(v16bf a, v16bf b, v8f c) {
  return __builtin_amdgcn_wmma_f32_16x16x32_bf16(false, a, false, b, (short)0, c, false, false);
}

__device__ __forceinline__ v8f v8f_zero() {
  v8f z;
#pragma unroll
  for (int i = 0; i < 8; ++i) z[i] = 0.f;
  return z;
}

// ---------------- conversion kernels ----------------

__global__ void cvt_bf16_kernel(const float* __restrict__ in, bf16* __restrict__ out, int n) {
  int i = blockIdx.x * blockDim.x + threadIdx.x;
  if (i < n) out[i] = (bf16)in[i];
}

// W stored [N=512][K=512] (out,in). Emit Wt[k][n] = W[n][k] in bf16, coalesced writes.
__global__ void transpose_cvt512_kernel(const float* __restrict__ in, bf16* __restrict__ out) {
  int idx = blockIdx.x * blockDim.x + threadIdx.x;   // 512*512
  int k = idx >> 9, n = idx & 511;
  out[idx] = (bf16)in[(size_t)n * 512 + k];
}

// ---------------- WMMA GEMM: C = A[M,K] @ Bt[K,512] + bias ----------------
// mode 0: bf16 head-split  out[((b*H+head)*T + t)*DK + dk],  val*scale
// mode 1: bf16 head-split transposed out[((b*H+head)*DK + dk)*T + t]
// mode 2: f32 row-major    out[row*512 + col]
__global__ void gemm_wmma_kernel(const bf16* __restrict__ A, const bf16* __restrict__ Bt,
                                 const float* __restrict__ bias, void* __restrict__ outp,
                                 int Kdim, int mode, float scale) {
  const int lane = threadIdx.x & 31;
  const int wave = threadIdx.x >> 5;
  const int m = lane & 15, h = lane >> 4;
  const int rowbase = blockIdx.x * 128 + wave * 16;
  const int nbase = blockIdx.y * 64;

  v8f acc[4];
#pragma unroll
  for (int nt = 0; nt < 4; ++nt) acc[nt] = v8f_zero();

  for (int kb = 0; kb < Kdim; kb += 32) {
    v16bf a = load_a(A + (size_t)rowbase * Kdim + kb, Kdim, m, h);
#pragma unroll
    for (int nt = 0; nt < 4; ++nt) {
      v16bf b = load_b(Bt + (size_t)kb * 512 + nbase + nt * 16, 512, m, h);
      acc[nt] = wmma_bf16(a, b, acc[nt]);
    }
  }

#pragma unroll
  for (int nt = 0; nt < 4; ++nt) {
    const int col = nbase + nt * 16 + m;
    const float bv = bias[col];
#pragma unroll
    for (int v = 0; v < 8; ++v) {
      const int row = rowbase + v + 8 * h;
      const float val = (acc[nt][v] + bv) * scale;
      if (mode == 2) {
        ((float*)outp)[(size_t)row * 512 + col] = val;
      } else {
        const int b_ = row >> 12;        // / T
        const int t  = row & (TT - 1);
        const int head = col >> 6;       // / DK
        const int dk = col & (TDK - 1);
        bf16* ob = (bf16*)outp;
        if (mode == 0)
          ob[(((size_t)b_ * TH + head) * TT + t) * TDK + dk] = (bf16)val;
        else
          ob[(((size_t)b_ * TH + head) * TDK + dk) * TT + t] = (bf16)val;
      }
    }
  }
}

// ---------------- flash attention ----------------
// grid (T/128, B*H), 256 threads. Wave w: 16 query rows. q pre-scaled by (1/8)*log2e.
// q: [BH][T][DK], kT: [BH][DK][T], v: [BH][T][DK]; ctx out bf16 [B*T][D].
__global__ void attn_wmma_kernel(const bf16* __restrict__ qg, const bf16* __restrict__ ktg,
                                 const bf16* __restrict__ vg, bf16* __restrict__ ctx) {
  __shared__ bf16 lds_p[8][16][32];   // per-wave P transpose staging (8 KB)
  const int lane = threadIdx.x & 31;
  const int wave = threadIdx.x >> 5;
  const int m = lane & 15, h = lane >> 4;
  const int bh = blockIdx.y;
  const int qrow = blockIdx.x * 128 + wave * 16;

  const bf16* qp  = qg  + (size_t)bh * TT * TDK;
  const bf16* ktp = ktg + (size_t)bh * TDK * TT;
  const bf16* vp  = vg  + (size_t)bh * TT * TDK;

  const v16bf qa0 = load_a(qp + (size_t)qrow * TDK + 0,  TDK, m, h);
  const v16bf qa1 = load_a(qp + (size_t)qrow * TDK + 32, TDK, m, h);

  v8f acc[4];
#pragma unroll
  for (int nt = 0; nt < 4; ++nt) acc[nt] = v8f_zero();
  v8f mrun, lrun;
#pragma unroll
  for (int i = 0; i < 8; ++i) { mrun[i] = -3.0e38f; lrun[i] = 0.f; }

  for (int kb = 0; kb < TT; kb += 32) {
    // S = Q @ K^T : B[d][n] rows contiguous in kT
    v16bf k00 = load_b(ktp + kb,                         TT, m, h); // d 0..31,  keys kb..+16
    v16bf k10 = load_b(ktp + (size_t)32 * TT + kb,       TT, m, h); // d 32..63
    v16bf k01 = load_b(ktp + kb + 16,                    TT, m, h); // keys kb+16..+32
    v16bf k11 = load_b(ktp + (size_t)32 * TT + kb + 16,  TT, m, h);
    v8f s0 = wmma_bf16(qa1, k10, wmma_bf16(qa0, k00, v8f_zero()));
    v8f s1 = wmma_bf16(qa1, k11, wmma_bf16(qa0, k01, v8f_zero()));

    // online softmax in log2 domain (scale*log2e folded into q)
    v8f mnew, alpha, p0, p1;
#pragma unroll
    for (int v = 0; v < 8; ++v) {
      float t = fmaxf(s0[v], s1[v]);
      t = fmaxf(t, __shfl_xor(t, 1, 32));
      t = fmaxf(t, __shfl_xor(t, 2, 32));
      t = fmaxf(t, __shfl_xor(t, 4, 32));
      t = fmaxf(t, __shfl_xor(t, 8, 32));
      mnew[v] = fmaxf(mrun[v], t);
    }
#pragma unroll
    for (int v = 0; v < 8; ++v) {
      alpha[v] = __builtin_amdgcn_exp2f(mrun[v] - mnew[v]);
      p0[v] = __builtin_amdgcn_exp2f(s0[v] - mnew[v]);
      p1[v] = __builtin_amdgcn_exp2f(s1[v] - mnew[v]);
      float rs = p0[v] + p1[v];
      rs += __shfl_xor(rs, 1, 32);
      rs += __shfl_xor(rs, 2, 32);
      rs += __shfl_xor(rs, 4, 32);
      rs += __shfl_xor(rs, 8, 32);
      lrun[v] = lrun[v] * alpha[v] + rs;
      mrun[v] = mnew[v];
    }
#pragma unroll
    for (int nt = 0; nt < 4; ++nt)
#pragma unroll
      for (int v = 0; v < 8; ++v) acc[nt][v] *= alpha[v];

    // transpose P (C/D layout, f32) -> A layout (bf16) via wave-private LDS
#pragma unroll
    for (int v = 0; v < 8; ++v) {
      const int row = v + 8 * h;
      lds_p[wave][row][m]      = (bf16)p0[v];
      lds_p[wave][row][16 + m] = (bf16)p1[v];
    }
    asm volatile("s_wait_dscnt 0" ::: "memory");   // wave-local: stores visible before reads
    v16bf pa = load_a(&lds_p[wave][0][0], 32, m, h);
    asm volatile("" ::: "memory");                 // keep reads before next-iter stores

    // ctx += P @ V : B rows are V rows (contiguous)
#pragma unroll
    for (int nt = 0; nt < 4; ++nt) {
      v16bf vb = load_b(vp + (size_t)kb * TDK + nt * 16, TDK, m, h);
      acc[nt] = wmma_bf16(pa, vb, acc[nt]);
    }
  }

  // epilogue: ctx[b, t, head*64+dk] = acc / l
  const int b_ = bh >> 3, head = bh & 7;
#pragma unroll
  for (int nt = 0; nt < 4; ++nt) {
#pragma unroll
    for (int v = 0; v < 8; ++v) {
      const int t = qrow + v + 8 * h;
      const int col = head * TDK + nt * 16 + m;
      ctx[((size_t)b_ * TT + t) * TD + col] = (bf16)(acc[nt][v] / lrun[v]);
    }
  }
}

// ---------------- launch ----------------

extern "C" void kernel_launch(void* const* d_in, const int* in_sizes, int n_in,
                              void* d_out, int out_size, void* d_ws, size_t ws_size,
                              hipStream_t stream) {
  const float* x  = (const float*)d_in[0];
  const float* Wq = (const float*)d_in[1];
  const float* bq = (const float*)d_in[2];
  const float* Wk = (const float*)d_in[3];
  const float* bk = (const float*)d_in[4];
  const float* Wv = (const float*)d_in[5];
  const float* bv = (const float*)d_in[6];
  const float* Wo = (const float*)d_in[7];
  const float* bo = (const float*)d_in[8];

  char* ws = (char*)d_ws;
  const size_t SZ_X  = (size_t)TB * TT * TD * sizeof(bf16);   // 8 MB
  const size_t SZ_W  = (size_t)TD * TD * sizeof(bf16);        // 0.5 MB
  bf16* xb   = (bf16*)(ws);
  bf16* wqT  = (bf16*)(ws + SZ_X);
  bf16* wkT  = (bf16*)(ws + SZ_X + SZ_W);
  bf16* wvT  = (bf16*)(ws + SZ_X + 2 * SZ_W);
  bf16* woT  = (bf16*)(ws + SZ_X + 3 * SZ_W);
  bf16* qbuf = (bf16*)(ws + SZ_X + 4 * SZ_W);
  bf16* ktbf = (bf16*)(ws + 2 * SZ_X + 4 * SZ_W);
  bf16* vbuf = (bf16*)(ws + 3 * SZ_X + 4 * SZ_W);
  bf16* ctxb = (bf16*)(ws + 4 * SZ_X + 4 * SZ_W);

  const int NX = TB * TT * TD;          // 4194304
  cvt_bf16_kernel<<<(NX + 255) / 256, 256, 0, stream>>>(x, xb, NX);
  transpose_cvt512_kernel<<<(TD * TD + 255) / 256, 256, 0, stream>>>(Wq, wqT);
  transpose_cvt512_kernel<<<(TD * TD + 255) / 256, 256, 0, stream>>>(Wk, wkT);
  transpose_cvt512_kernel<<<(TD * TD + 255) / 256, 256, 0, stream>>>(Wv, wvT);
  transpose_cvt512_kernel<<<(TD * TD + 255) / 256, 256, 0, stream>>>(Wo, woT);

  dim3 ggrid(TB * TT / 128, TD / 64);   // (64, 8)
  const float qscale = 0.125f * 1.44269504088896340736f;  // 1/sqrt(DK) * log2(e)
  gemm_wmma_kernel<<<ggrid, 256, 0, stream>>>(xb, wqT, bq, qbuf, TD, 0, qscale);
  gemm_wmma_kernel<<<ggrid, 256, 0, stream>>>(xb, wkT, bk, ktbf, TD, 1, 1.0f);
  gemm_wmma_kernel<<<ggrid, 256, 0, stream>>>(xb, wvT, bv, vbuf, TD, 0, 1.0f);

  attn_wmma_kernel<<<dim3(TT / 128, TB * TH), 256, 0, stream>>>(qbuf, ktbf, vbuf, ctxb);

  gemm_wmma_kernel<<<ggrid, 256, 0, stream>>>(ctxb, woT, bo, d_out, TD, 2, 1.0f);
}
